// TurnInterface_89137751261848
// MI455X (gfx1250) — matrix operand order repaired and checked by reading, not compiled, not run on hardware
//
#include <hip/hip_runtime.h>
#include <math.h>
#include <stdint.h>

// ---------------- problem constants ----------------
#define HN     16
#define DKC    64
#define DVC    64
#define PBATCH 4
#define TT     4096
#define DD     2048
#define NPHASE 6
#define TEMPC  5.0f
#define GAMMAC 0.9f

typedef _Float16 half_t;
typedef __attribute__((ext_vector_type(16))) _Float16 v16h;
typedef __attribute__((ext_vector_type(8)))  _Float16 v8h;
typedef __attribute__((ext_vector_type(4)))  _Float16 v4h;
typedef __attribute__((ext_vector_type(8)))  float    v8f;
typedef __attribute__((ext_vector_type(4)))  unsigned int u32x4;
typedef __attribute__((ext_vector_type(8)))  int      i32x8;
typedef __attribute__((ext_vector_type(4)))  int      i32x4;

#define AS1 __attribute__((address_space(1)))
#define AS3 __attribute__((address_space(3)))

// ---------------- feature probes ----------------
#if defined(__has_builtin)
#  if __has_builtin(__builtin_amdgcn_global_load_async_to_lds_b128)
#    define HAS_ASYNC_LDS 1
#  endif
#  if __has_builtin(__builtin_amdgcn_tensor_load_to_lds)
#    define HAS_TDM 1
#  endif
#  if __has_builtin(__builtin_amdgcn_s_wait_asynccnt)
#    define HAS_ACNT 1
#  endif
#  if __has_builtin(__builtin_amdgcn_s_wait_tensorcnt)
#    define HAS_TCNT 1
#  endif
#endif
#ifndef HAS_ASYNC_LDS
#define HAS_ASYNC_LDS 0
#endif
#ifndef HAS_TDM
#define HAS_TDM 0
#endif
#ifndef HAS_ACNT
#define HAS_ACNT 0
#endif
#ifndef HAS_TCNT
#define HAS_TCNT 0
#endif

// ---------------- WMMA core ----------------
__device__ __forceinline__ v8f wmma16(v16h a, v16h b, v8f c) {
  return __builtin_amdgcn_wmma_f32_16x16x32_f16(
      false, a, false, b, (short)0, c, false, false);
}

// A-matrix 16x32 f16 fragment: lane L holds row M=L&15,
// K in {0..7,16..23} (L<16) or {8..15,24..31} (L>=16).
__device__ __forceinline__ v16h load_a(const half_t* base, int ld, int lane) {
  const int m  = lane & 15;
  const int kb = (lane < 16) ? 0 : 8;
  const half_t* p = base + (size_t)m * ld + kb;
  v8h lo = *(const v8h*)(p);
  v8h hi = *(const v8h*)(p + 16);
  v16h a;
#pragma unroll
  for (int i = 0; i < 8; ++i) { a[i] = lo[i]; a[i + 8] = hi[i]; }
  return a;
}

// B-matrix 32x16 f16 fragment: lane L holds column N=L&15, K {0..7,16..23} / {8..15,24..31}.
// Source stored row-major N x K.
__device__ __forceinline__ v16h load_b(const half_t* base, int ld, int lane) {
  const int n  = lane & 15;
  const int kb = (lane < 16) ? 0 : 8;
  const half_t* p = base + (size_t)n * ld + kb;
  v8h lo = *(const v8h*)(p);
  v8h hi = *(const v8h*)(p + 16);
  v16h b;
#pragma unroll
  for (int i = 0; i < 8; ++i) { b[i] = lo[i]; b[i + 8] = hi[i]; }
  return b;
}

__device__ __forceinline__ float redux16(float x) {
  x += __shfl_xor(x, 1);
  x += __shfl_xor(x, 2);
  x += __shfl_xor(x, 4);
  x += __shfl_xor(x, 8);
  return x;
}

// ---------------- async copy global -> LDS (16B per lane) ----------------
__device__ __forceinline__ void async_b128(const half_t* g, half_t* l) {
#if HAS_ASYNC_LDS
  __builtin_amdgcn_global_load_async_to_lds_b128(
      (AS1 i32x4*)(unsigned long long)(uintptr_t)g,
      (AS3 i32x4*)(unsigned)(uintptr_t)l, 0, 0);
#else
  *(v8h*)l = *(const v8h*)g;
#endif
}

__device__ __forceinline__ void wait_async_2or0(bool more) {
#if HAS_ASYNC_LDS
#  if HAS_ACNT
  if (more) __builtin_amdgcn_s_wait_asynccnt(2);
  else      __builtin_amdgcn_s_wait_asynccnt(0);
#  else
  if (more) asm volatile("s_wait_asynccnt 0x2" ::: "memory");
  else      asm volatile("s_wait_asynccnt 0x0" ::: "memory");
#  endif
#else
  (void)more;
#endif
}

__device__ __forceinline__ void wait_async_1or0(bool more) {
#if HAS_ASYNC_LDS
#  if HAS_ACNT
  if (more) __builtin_amdgcn_s_wait_asynccnt(1);
  else      __builtin_amdgcn_s_wait_asynccnt(0);
#  else
  if (more) asm volatile("s_wait_asynccnt 0x1" ::: "memory");
  else      asm volatile("s_wait_asynccnt 0x0" ::: "memory");
#  endif
#else
  (void)more;
#endif
}

// stage 64 rows x 32 halves (4 KB) with 128 threads, 2 x b128 per thread
__device__ __forceinline__ void stage64(const half_t* g, half_t* l, int ld, int tid) {
#pragma unroll
  for (int i = 0; i < 2; ++i) {
    int c = (tid + i * 128) * 8;
    int r = c >> 5, k = c & 31;
    async_b128(g + (size_t)r * ld + k, l + (r << 5) + k);
  }
}

// ---------------- Tensor Data Mover: 2D f16 tile -> LDS ----------------
#if HAS_TDM
__device__ __forceinline__ void tdm_load_2d(unsigned lds_addr, unsigned long long gaddr,
                                            unsigned tile_rows, unsigned tile_cols,
                                            unsigned stride_elems, unsigned tensor_cols,
                                            unsigned tensor_rows) {
  u32x4 g0;
  g0[0] = 1u;                                    // count=1, user descriptor
  g0[1] = lds_addr;                              // LDS byte address
  g0[2] = (unsigned)gaddr;                       // global addr [31:0]
  g0[3] = (unsigned)(gaddr >> 32) | (2u << 30);  // addr [56:32] | type=2 (image)
  i32x8 g1;
  g1[0] = (int)(1u << 16);                                            // data_size=2B
  g1[1] = (int)((tensor_cols & 0xFFFFu) << 16);                       // tensor_dim0 lo
  g1[2] = (int)((tensor_cols >> 16) | ((tensor_rows & 0xFFFFu) << 16));// td0 hi | td1 lo
  g1[3] = (int)((tensor_rows >> 16) | (tile_cols << 16));             // td1 hi | tile_dim0
  g1[4] = (int)tile_rows;                                             // tile_dim1 (dim2=0)
  g1[5] = (int)stride_elems;                                          // dim0 stride lo32
  g1[6] = 0;                                                          // stride hi | dim1 stride
  g1[7] = 0;
  i32x4 z4 = {0, 0, 0, 0};
#  if __clang_major__ >= 23
  i32x8 z8 = {0, 0, 0, 0, 0, 0, 0, 0};
  __builtin_amdgcn_tensor_load_to_lds(g0, g1, z4, z4, z8, 0);
#  else
  __builtin_amdgcn_tensor_load_to_lds(g0, g1, z4, z4, 0);
#  endif
}
#endif

__device__ __forceinline__ void wait_tensor_1or0(bool more) {
#if HAS_TCNT
  if (more) __builtin_amdgcn_s_wait_tensorcnt(1);
  else      __builtin_amdgcn_s_wait_tensorcnt(0);
#else
  if (more) asm volatile("s_wait_tensorcnt 0x1" ::: "memory");
  else      asm volatile("s_wait_tensorcnt 0x0" ::: "memory");
#endif
}

// stage 32 rows x 32 halves (2 KB): TDM if present, else per-lane async/sync
__device__ __forceinline__ void stage32(const half_t* g, half_t* l, int ld, int tid) {
#if HAS_TDM
  if (tid < 32)
    tdm_load_2d((unsigned)(uintptr_t)l, (unsigned long long)(uintptr_t)g,
                32u, 32u, (unsigned)ld, (unsigned)ld, (unsigned)TT);
#else
  int c = tid * 8;
  int r = c >> 5, k = c & 31;
  async_b128(g + (size_t)r * ld + k, l + (r << 5) + k);
#endif
}

__device__ __forceinline__ void stage32_wait(bool more, int tid) {
#if HAS_TDM
  if (tid < 32) wait_tensor_1or0(more);
#else
  wait_async_1or0(more);
#endif
}

// ---------------- f32 -> f16 converter ----------------
__global__ void k_f32_to_f16(const float* __restrict__ src,
                             half_t* __restrict__ dst, int n) {
  int i = (blockIdx.x * blockDim.x + threadIdx.x) * 4;
  const int stride = gridDim.x * blockDim.x * 4;
  for (; i + 3 < n; i += stride) {
    float4 f = *(const float4*)(src + i);
    v4h h;
    h[0] = (half_t)f.x; h[1] = (half_t)f.y;
    h[2] = (half_t)f.z; h[3] = (half_t)f.w;
    *(v4h*)(dst + i) = h;
  }
}

// ---------------- phase-rotated reads helper ----------------
__device__ __forceinline__ void qrot_reads(const v8f* q, int p, float f0, float f1,
                                           half_t* st, const half_t* Wt,
                                           int lane, int m_hi, int ncol, v8f* rd) {
  const float a0 = (float)p * f0, a1 = (float)p * f1;
  const float c0 = __cosf(a0), s0 = __sinf(a0);
  const float c1 = __cosf(a1), s1 = __sinf(a1);
#pragma unroll
  for (int nt = 0; nt < 4; ++nt) {
    const float cc  = (nt & 1) ? c1 : c0;
    const float ss  = (nt & 1) ? s1 : s0;
    const float sgn = (nt < 2) ? -1.f : 1.f;
    const int dcol = nt * 16 + ncol;
#pragma unroll
    for (int r = 0; r < 8; ++r) {
      float v = q[nt][r] * cc + sgn * q[nt ^ 2][r] * ss;
      st[(m_hi + r) * 64 + dcol] = (half_t)v;
    }
  }
  __builtin_amdgcn_wave_barrier();
#pragma unroll
  for (int kk = 0; kk < 2; ++kk) {
    v16h a = load_a(st + kk * 32, 64, lane);
#pragma unroll
    for (int vt = 0; vt < 4; ++vt) {
      v16h bm = load_b(Wt + (size_t)(vt * 16) * 64 + kk * 32, 64, lane);
      rd[vt] = wmma16(a, bm, rd[vt]);
    }
  }
  __builtin_amdgcn_wave_barrier();
}

// ---------------- read path ----------------
__global__ __launch_bounds__(128) void k_read(const half_t* __restrict__ Xh,
                                              const half_t* __restrict__ Wq,
                                              const float* __restrict__ Wturn,
                                              half_t* __restrict__ Mrg) {
  __shared__ __attribute__((aligned(32))) half_t Wt[64 * 64];
  __shared__ __attribute__((aligned(32))) half_t stage[4][16 * 64];
  __shared__ __attribute__((aligned(32))) half_t Xs[2][64 * 32];

  const int tid  = threadIdx.x;
  const int lane = tid & 31;
  const int wave = tid >> 5;
  const int b = blockIdx.x >> 4;
  const int h = blockIdx.x & 15;
  const int t0 = blockIdx.y * 64 + wave * 16;

  {  // cooperative W_turn[b,h] -> f16 LDS
    const float* wsrc = Wturn + (size_t)(b * HN + h) * 64 * 64;
    for (int i = tid; i < 4096; i += 128) Wt[i] = (half_t)wsrc[i];
  }
  __syncthreads();

  const int m_hi = (lane >> 4) << 3;
  const int ncol = lane & 15;

  // ---- Q tile (16 x 64 per wave), A staged via async LDS double-buffer ----
  v8f q[4] = {};
  const half_t* Ag = Xh + (size_t)(b * TT + blockIdx.y * 64) * DD;
  const half_t* Bb = Wq + (size_t)(h * 64) * DD;
  stage64(Ag, Xs[0], DD, tid);
  for (int kk = 0; kk < DD / 32; ++kk) {
    const int cur = kk & 1;
    const bool more = (kk + 1 < DD / 32);
    if (more) stage64(Ag + (kk + 1) * 32, Xs[cur ^ 1], DD, tid);
    wait_async_2or0(more);
    __syncthreads();
    v16h a = load_a(Xs[cur] + wave * 16 * 32, 32, lane);
#pragma unroll
    for (int nt = 0; nt < 4; ++nt) {
      v16h bm = load_b(Bb + (size_t)(nt * 16) * DD + kk * 32, DD, lane);
      q[nt] = wmma16(a, bm, q[nt]);
    }
    __syncthreads();
  }

  // row-normalize q
#pragma unroll
  for (int r = 0; r < 8; ++r) {
    float s = 0.f;
#pragma unroll
    for (int nt = 0; nt < 4; ++nt) { float x = q[nt][r]; s += x * x; }
    s = redux16(s);
    float inv = 1.0f / fmaxf(__builtin_sqrtf(s), 1e-12f);
#pragma unroll
    for (int nt = 0; nt < 4; ++nt) q[nt][r] *= inv;
  }

  const float lf = __logf(10000.0f);
  const float f0 = __expf(-(float)(ncol)*0.03125f * lf);
  const float f1 = __expf(-(float)(ncol + 16) * 0.03125f * lf);

  half_t* st = stage[wave];

  // pass 1: phase scores
  float sc[NPHASE][8];
#pragma unroll
  for (int p = 0; p < NPHASE; ++p) {
    v8f rd[4] = {};
    qrot_reads(q, p, f0, f1, st, Wt, lane, m_hi, ncol, rd);
#pragma unroll
    for (int r = 0; r < 8; ++r) {
      float s = 0.f;
#pragma unroll
      for (int vt = 0; vt < 4; ++vt) { float x = rd[vt][r]; s += x * x; }
      s = redux16(s);
      sc[p][r] = __builtin_sqrtf(s) * 0.125f * TEMPC;
    }
  }

  // softmax over phases (in place)
#pragma unroll
  for (int r = 0; r < 8; ++r) {
    float mx = sc[0][r];
#pragma unroll
    for (int p = 1; p < NPHASE; ++p) mx = fmaxf(mx, sc[p][r]);
    float sum = 0.f;
#pragma unroll
    for (int p = 0; p < NPHASE; ++p) { float e = __expf(sc[p][r] - mx); sc[p][r] = e; sum += e; }
    float inv = 1.0f / sum;
#pragma unroll
    for (int p = 0; p < NPHASE; ++p) sc[p][r] *= inv;
  }

  // pass 2: recompute reads, weighted accumulate
  v8f racc[4] = {};
#pragma unroll
  for (int p = 0; p < NPHASE; ++p) {
    v8f rd[4] = {};
    qrot_reads(q, p, f0, f1, st, Wt, lane, m_hi, ncol, rd);
#pragma unroll
    for (int vt = 0; vt < 4; ++vt)
#pragma unroll
      for (int r = 0; r < 8; ++r) racc[vt][r] += sc[p][r] * rd[vt][r];
  }

  half_t* mout = Mrg + (size_t)(b * TT + t0) * (HN * DVC) + h * 64;
#pragma unroll
  for (int nt = 0; nt < 4; ++nt)
#pragma unroll
    for (int r = 0; r < 8; ++r)
      mout[(size_t)(m_hi + r) * (HN * DVC) + nt * 16 + ncol] = (half_t)racc[nt][r];
}

// ---------------- out = hidden + sigmoid(rs) * merged @ o_w^T ----------------
__global__ __launch_bounds__(128) void k_out(const half_t* __restrict__ Mrg,
                                             const half_t* __restrict__ Wo,
                                             const float* __restrict__ hidden,
                                             const float* __restrict__ rscale,
                                             float* __restrict__ out) {
  __shared__ __attribute__((aligned(32))) half_t Xs[2][64 * 32];

  const int tid  = threadIdx.x;
  const int lane = tid & 31;
  const int wave = tid >> 5;
  const int row0 = blockIdx.x * 64;
  const int col0 = blockIdx.y * 64;
  const int KDIM = HN * DVC;  // 1024

  v8f acc[4] = {};
  const half_t* Ag = Mrg + (size_t)row0 * KDIM;
  const half_t* Bb = Wo + (size_t)col0 * KDIM;
  stage64(Ag, Xs[0], KDIM, tid);
  for (int kk = 0; kk < KDIM / 32; ++kk) {
    const int cur = kk & 1;
    const bool more = (kk + 1 < KDIM / 32);
    if (more) stage64(Ag + (kk + 1) * 32, Xs[cur ^ 1], KDIM, tid);
    wait_async_2or0(more);
    __syncthreads();
    v16h a = load_a(Xs[cur] + wave * 16 * 32, 32, lane);
#pragma unroll
    for (int nt = 0; nt < 4; ++nt) {
      v16h bm = load_b(Bb + (size_t)(nt * 16) * KDIM + kk * 32, KDIM, lane);
      acc[nt] = wmma16(a, bm, acc[nt]);
    }
    __syncthreads();
  }

  const float sig = 1.0f / (1.0f + __expf(-rscale[0]));
  const int m_hi = (lane >> 4) << 3;
  const int ncol = lane & 15;
#pragma unroll
  for (int nt = 0; nt < 4; ++nt)
#pragma unroll
    for (int r = 0; r < 8; ++r) {
      size_t idx = (size_t)(row0 + wave * 16 + m_hi + r) * DD + col0 + nt * 16 + ncol;
      out[idx] = hidden[idx] + sig * acc[nt][r];
    }
}

// ---------------- W_new = gamma * age(W_turn) + (V^T K)/T ----------------
__global__ __launch_bounds__(128) void k_wnew(const half_t* __restrict__ Xh,
                                              const half_t* __restrict__ Wk,
                                              const half_t* __restrict__ Wv,
                                              const float* __restrict__ Wturn,
                                              float* __restrict__ Wnew) {
  __shared__ __attribute__((aligned(32))) half_t kbt[64 * 32];  // [d][t]
  __shared__ __attribute__((aligned(32))) half_t vbt[64 * 32];  // [v][t]
  __shared__ __attribute__((aligned(32))) half_t Xs[2][32 * 32];

  const int tid  = threadIdx.x;
  const int lane = tid & 31;
  const int wave = tid >> 5;
  const int b = blockIdx.x >> 4;
  const int h = blockIdx.x & 15;
  const int m_hi = (lane >> 4) << 3;
  const int ncol = lane & 15;

  const bool isK = (wave < 2);
  const int mt = wave & 1;
  const half_t* Bb = (isK ? Wk : Wv) + (size_t)(h * 64) * DD;
  half_t* tbuf = isK ? kbt : vbt;

  v8f res[4] = {};
  for (int tc = 0; tc < TT / 32; ++tc) {
    const int t0 = tc * 32;

    // 16x64 slice of K or V: X tile (32x32) staged via TDM double-buffer
    v8f acc[4] = {};
    const half_t* Ag = Xh + (size_t)(b * TT + t0) * DD;
    stage32(Ag, Xs[0], DD, tid);
    for (int kk = 0; kk < DD / 32; ++kk) {
      const int cur = kk & 1;
      const bool more = (kk + 1 < DD / 32);
      if (more) stage32(Ag + (kk + 1) * 32, Xs[cur ^ 1], DD, tid);
      stage32_wait(more, tid);
      __syncthreads();
      v16h a = load_a(Xs[cur] + mt * 16 * 32, 32, lane);
#pragma unroll
      for (int nt = 0; nt < 4; ++nt) {
        v16h bm = load_b(Bb + (size_t)(nt * 16) * DD + kk * 32, DD, lane);
        acc[nt] = wmma16(a, bm, acc[nt]);
      }
      __syncthreads();
    }

    if (isK) {  // normalize k rows over dk=64
#pragma unroll
      for (int r = 0; r < 8; ++r) {
        float s = 0.f;
#pragma unroll
        for (int nt = 0; nt < 4; ++nt) { float x = acc[nt][r]; s += x * x; }
        s = redux16(s);
        float inv = 1.0f / fmaxf(__builtin_sqrtf(s), 1e-12f);
#pragma unroll
        for (int nt = 0; nt < 4; ++nt) acc[nt][r] *= inv;
      }
    }

    // store transposed: tbuf[d_or_v][t_local]
#pragma unroll
    for (int nt = 0; nt < 4; ++nt)
#pragma unroll
      for (int r = 0; r < 8; ++r)
        tbuf[(nt * 16 + ncol) * 32 + mt * 16 + m_hi + r] = (half_t)acc[nt][r];
    __syncthreads();

    // residual[v][d] += sum_t V[t][v] * K[t][d]
    v16h a = load_a(vbt + (size_t)(wave * 16) * 32, 32, lane);
#pragma unroll
    for (int ct = 0; ct < 4; ++ct) {
      v16h bm = load_b(kbt + (size_t)(ct * 16) * 32, 32, lane);
      res[ct] = wmma16(a, bm, res[ct]);
    }
    __syncthreads();
  }

  // epilogue: 1-step RoPE aging + gamma + residual/T
  const float lf = __logf(10000.0f);
  const float invT = 1.0f / (float)TT;
  const float* Wsrc = Wturn + (size_t)(b * HN + h) * 64 * 64;
  float* Wdst = Wnew + (size_t)(b * HN + h) * 64 * 64;
#pragma unroll
  for (int ct = 0; ct < 4; ++ct) {
    const int d = ct * 16 + ncol;
    const float fr = __expf(-(float)(d & 31) * 0.03125f * lf);
    const float c1 = __cosf(fr), s1 = __sinf(fr);
    const float sgn = (d < 32) ? -1.f : 1.f;
    const int dp = d ^ 32;
#pragma unroll
    for (int r = 0; r < 8; ++r) {
      const int vrow = wave * 16 + m_hi + r;
      float w = Wsrc[(size_t)vrow * 64 + d];
      float wp = Wsrc[(size_t)vrow * 64 + dp];
      float aged = w * c1 + sgn * wp * s1;
      Wdst[(size_t)vrow * 64 + d] = GAMMAC * aged + res[ct][r] * invT;
    }
  }
}

// ---------------- host side ----------------
extern "C" void kernel_launch(void* const* d_in, const int* in_sizes, int n_in,
                              void* d_out, int out_size, void* d_ws, size_t ws_size,
                              hipStream_t stream) {
  (void)in_sizes; (void)n_in; (void)out_size; (void)ws_size;
  const float* hidden = (const float*)d_in[0];
  const float* Wturn  = (const float*)d_in[1];
  const float* q_w    = (const float*)d_in[2];
  const float* o_w    = (const float*)d_in[3];
  const float* k_w    = (const float*)d_in[4];
  const float* v_w    = (const float*)d_in[5];
  const float* rscale = (const float*)d_in[6];

  float* out  = (float*)d_out;
  float* Wnew = out + (size_t)PBATCH * TT * DD;

  char* ws = (char*)d_ws;
  half_t* Xh  = (half_t*)(ws);
  half_t* Mrg = (half_t*)(ws + (size_t)(64u << 20));
  half_t* Wq  = (half_t*)(ws + (size_t)(96u << 20));
  half_t* Wk  = (half_t*)(ws + (size_t)(100u << 20));
  half_t* Wv  = (half_t*)(ws + (size_t)(104u << 20));
  half_t* Wo  = (half_t*)(ws + (size_t)(108u << 20));

  const int nX = PBATCH * TT * DD;
  const int nW = HN * DKC * DD;

  k_f32_to_f16<<<2048, 256, 0, stream>>>(hidden, Xh, nX);
  k_f32_to_f16<<<512, 256, 0, stream>>>(q_w, Wq, nW);
  k_f32_to_f16<<<512, 256, 0, stream>>>(k_w, Wk, nW);
  k_f32_to_f16<<<512, 256, 0, stream>>>(v_w, Wv, nW);
  k_f32_to_f16<<<512, 256, 0, stream>>>(o_w, Wo, nW);

  k_read<<<dim3(PBATCH * HN, TT / 64), 128, 0, stream>>>(Xh, Wq, Wturn, Mrg);
  k_out<<<dim3((PBATCH * TT) / 64, DD / 64), 128, 0, stream>>>(Mrg, Wo, hidden, rscale, out);
  k_wnew<<<PBATCH * HN, 128, 0, stream>>>(Xh, Wk, Wv, Wturn, Wnew);
}
